// MiniResNetJoint_7181185319222
// MI455X (gfx1250) — matrix operand order repaired and checked by reading, not compiled, not run on hardware
//
#include <hip/hip_runtime.h>
#include <hip/hip_bf16.h>
#include <math.h>

// ---------------------------------------------------------------------------
// MiniResNetJoint forward for MI455X (gfx1250, wave32, WMMA).
// Convs + linear heads run as implicit GEMM on v_wmma_f32_16x16x32_bf16.
// ---------------------------------------------------------------------------

typedef float  v8f   __attribute__((ext_vector_type(8)));
typedef __bf16 v16bf __attribute__((ext_vector_type(16)));
typedef __bf16 v8bf  __attribute__((ext_vector_type(8)));

union Frag16 { v16bf v; v8bf h[2]; };

#define BATCH 128
#define NCLS  3755
#define NRAD  214
#define NSTRUCT 13
#define NSC   30
#define NST   6

// ---------------------------------------------------------------------------
// Weight convert: f32 -> bf16, layout preserved ([Cout][K] row-major, which is
// exactly the natural flatten of [Cout][Cin][ks][ks]).
// ---------------------------------------------------------------------------
__global__ void k_cvtw(const float* __restrict__ w, __bf16* __restrict__ out,
                       int total) {
  int i = blockIdx.x * 256 + threadIdx.x;
  if (i < total) out[i] = (__bf16)w[i];
}

// ---------------------------------------------------------------------------
// Stem conv: 1->64 channels, 3x3 pad 1, 64x64. K=9: direct (memory-bound).
// ---------------------------------------------------------------------------
__global__ void k_stem(const float* __restrict__ x, const float* __restrict__ w,
                       float* __restrict__ out) {
  int i = blockIdx.x * 256 + threadIdx.x;      // (n,co,h,w) over 128*64*64*64
  int wp = i & 63, h = (i >> 6) & 63, co = (i >> 12) & 63, n = i >> 18;
  const float* xp = x + (size_t)n * 4096;
  float acc = 0.f;
#pragma unroll
  for (int r = 0; r < 3; ++r) {
    int ih = h + r - 1;
    if ((unsigned)ih >= 64u) continue;
#pragma unroll
    for (int s = 0; s < 3; ++s) {
      int iw = wp + s - 1;
      if ((unsigned)iw >= 64u) continue;
      acc += xp[ih * 64 + iw] * w[co * 9 + r * 3 + s];
    }
  }
  out[i] = acc;
}

// ---------------------------------------------------------------------------
// BatchNorm (training-mode batch stats) + apply (+residual, +ReLU).
// All C and HW are powers of two -> shift/mask addressing.
// ---------------------------------------------------------------------------
__global__ void k_bn_stats(const float* __restrict__ in, float* __restrict__ stats,
                           int lc, int hw_s) {
  __shared__ float s1[256], s2[256];
  int c = blockIdx.x, tid = threadIdx.x;
  const int HW = 1 << hw_s;
  const int total = BATCH << hw_s;
  float a = 0.f, b = 0.f;
  for (int i = tid; i < total; i += 256) {
    int n = i >> hw_s, r = i & (HW - 1);
    float v = in[(size_t)(((n << lc) + c) << hw_s) + r];
    a += v; b += v * v;
  }
  s1[tid] = a; s2[tid] = b; __syncthreads();
  for (int s = 128; s > 0; s >>= 1) {
    if (tid < s) { s1[tid] += s1[tid + s]; s2[tid] += s2[tid + s]; }
    __syncthreads();
  }
  if (tid == 0) {
    float m = s1[0] / (float)total;
    float var = s2[0] / (float)total - m * m;
    stats[2 * c] = m;
    stats[2 * c + 1] = rsqrtf(fmaxf(var, 0.f) + 1e-5f);
  }
}

__global__ void k_bn_apply(const float* __restrict__ in, const float* __restrict__ res,
                           float* __restrict__ out, const float* __restrict__ stats,
                           const float* __restrict__ g, const float* __restrict__ b,
                           int lc, int hw_s, int total, int relu) {
  int i = blockIdx.x * 256 + threadIdx.x;
  if (i >= total) return;
  int c = (i >> hw_s) & ((1 << lc) - 1);
  float y = (in[i] - stats[2 * c]) * stats[2 * c + 1] * g[c] + b[c];
  if (res) y += res[i];
  if (relu) y = fmaxf(y, 0.f);
  out[i] = y;
}

// ---------------------------------------------------------------------------
// 2x2 max pool, stride 2 (square, power-of-two dims).
// ---------------------------------------------------------------------------
__global__ void k_maxpool(const float* __restrict__ in, float* __restrict__ out,
                          int lw, int lc, int total_out) {
  int i = blockIdx.x * 256 + threadIdx.x;
  if (i >= total_out) return;
  const int lwo = lw - 1;
  int ow = i & ((1 << lwo) - 1);
  int oh = (i >> lwo) & ((1 << lwo) - 1);
  int c  = (i >> (2 * lwo)) & ((1 << lc) - 1);
  int n  = i >> (2 * lwo + lc);
  const int W = 1 << lw;
  const float* p = in + (size_t)((((n << lc) + c) << (2 * lw)) + (oh << (lw + 1)) + (ow << 1));
  out[i] = fmaxf(fmaxf(p[0], p[1]), fmaxf(p[W], p[W + 1]));
}

// ---------------------------------------------------------------------------
// Global average pool (128,256,8,8) -> (128,256)
// ---------------------------------------------------------------------------
__global__ void k_gap(const float* __restrict__ in, float* __restrict__ out) {
  int i = blockIdx.x * 256 + threadIdx.x;
  if (i >= BATCH * 256) return;
  const float* p = in + (size_t)i * 64;
  float s = 0.f;
#pragma unroll
  for (int j = 0; j < 64; ++j) s += p[j];
  out[i] = s * (1.f / 64.f);
}

// ---------------------------------------------------------------------------
// Implicit-GEMM conv / linear on WMMA bf16 (f32 accumulate).
//   out[p, co] = sum_k A[p,k] * B[k,co] (+bias[co]),  p=(n,h,w), k=c*ks^2+tap
// Workgroup = 128 threads (4 waves). Tile: 128 pixels x 64 out-channels.
// Each wave: 2 A-frags (32 pixel rows) x 4 B-frags (64 cols) = 8 WMMA/K-step.
// A tile staged in LDS so each lane's two 8-element K-runs are contiguous
// (ds_load_b128); B fragments read directly from bf16 weights in natural
// [Cout][K] order (two global_load_b128 per lane, L2-resident).
// ---------------------------------------------------------------------------
template <int KS>
__global__ __launch_bounds__(128) void k_gemm_wmma(
    const float* __restrict__ in, const __bf16* __restrict__ wBT,  // [Cout][K]
    const float* __restrict__ bias, float* __restrict__ out,
    int Cin, int hw_s, int w_s, int Cout) {
  __shared__ unsigned As32[128][20];   // 128 pixels x 32 bf16 K (pitch 80B)

  const int tid = threadIdx.x;
  const int lane = tid & 31, wv = tid >> 5;
  const int HWl = 1 << hw_s;
  const int W = 1 << w_s;
  const int H = HWl >> w_s;
  const int pbase = blockIdx.x * 128;
  const int cobase = blockIdx.y * 64;
  const int K = Cin * (KS * KS);

  v8f acc[2][4] = {};

  for (int k0 = 0; k0 < K; k0 += 32) {
    // ---- stage A tile (im2col f32 -> packed bf16 pairs) ----
    for (int idx = tid; idx < 128 * 16; idx += 128) {
      int pl = idx >> 4, kp = idx & 15;
      int gp = pbase + pl;
      int n = gp >> hw_s, rem = gp & (HWl - 1);
      int h = rem >> w_s, w = rem & (W - 1);
      unsigned pack = 0;
#pragma unroll
      for (int u = 0; u < 2; ++u) {
        int k = k0 + kp * 2 + u;
        int c, dh, dw;
        if (KS == 1) { c = k; dh = 0; dw = 0; }
        else { c = k / 9; int t9 = k - c * 9; int t3 = t9 / 3;
               dh = t3 - 1; dw = t9 - t3 * 3 - 1; }
        int ih = h + dh, iw = w + dw;
        float v = 0.f;
        if ((unsigned)ih < (unsigned)H && (unsigned)iw < (unsigned)W)
          v = in[(((size_t)n * Cin + c) << hw_s) + (ih << w_s) + iw];
        unsigned short us = __builtin_bit_cast(unsigned short, (__bf16)v);
        pack |= (unsigned)us << (16 * u);
      }
      As32[pl][kp] = pack;
    }
    __syncthreads();

    if (tid == 0 && k0 + 32 < K)   // gfx1250 global_prefetch_b8 of next W slice
      __builtin_prefetch(&wBT[(size_t)cobase * K + k0 + 32], 0, 1);

    // ---- B fragments straight from global (two b128 per lane per frag) ----
    Frag16 bfr[4];
#pragma unroll
    for (int t = 0; t < 4; ++t) {
      int col = cobase + t * 16 + (lane & 15);
      if (col >= Cout) col = Cout - 1;            // clamp; results never stored
      const __bf16* bp = wBT + (size_t)col * K + k0 + ((lane < 16) ? 0 : 8);
      bfr[t].h[0] = *(const v8bf*)bp;
      bfr[t].h[1] = *(const v8bf*)(bp + 16);
    }
    // ---- A fragments from LDS (two b128 per lane) + 8 WMMAs ----
#pragma unroll
    for (int fr = 0; fr < 2; ++fr) {
      Frag16 a;
      const int rowi = wv * 32 + fr * 16 + (lane & 15);
      const __bf16* ap = (const __bf16*)&As32[rowi][0] + ((lane < 16) ? 0 : 8);
      a.h[0] = *(const v8bf*)ap;          // K = kb .. kb+7
      a.h[1] = *(const v8bf*)(ap + 16);   // K = kb+16 .. kb+23
#pragma unroll
      for (int t = 0; t < 4; ++t)
        acc[fr][t] = __builtin_amdgcn_wmma_f32_16x16x32_bf16(
            false, a.v, false, bfr[t].v, (short)0, acc[fr][t], false, false);
    }
    __syncthreads();
  }

  // ---- store (C/D layout: lane<16 -> M=r, N=lane; lane>=16 -> M=r+8) ----
#pragma unroll
  for (int t = 0; t < 4; ++t) {
    int co = cobase + t * 16 + (lane & 15);
    if (co >= Cout) continue;
    float badd = bias ? bias[co] : 0.f;
#pragma unroll
    for (int fr = 0; fr < 2; ++fr) {
#pragma unroll
      for (int r = 0; r < 8; ++r) {
        int m = wv * 32 + fr * 16 + ((lane < 16) ? r : r + 8);
        int gp = pbase + m;
        int n = gp >> hw_s, rem = gp & (HWl - 1);
        out[(((size_t)n * Cout + co) << hw_s) + rem] = acc[fr][t][r] + badd;
      }
    }
  }
}

// ---------------------------------------------------------------------------
// Symbolic reranker: one workgroup (128 threads) per batch row.
// ---------------------------------------------------------------------------
__global__ void k_rerank(const float* __restrict__ char_l, const float* __restrict__ rad_l,
                         const float* __restrict__ sc_l, const float* __restrict__ st_l,
                         const float* __restrict__ str_l,
                         const int* __restrict__ radical_mask,
                         const int* __restrict__ structure_label,
                         const int* __restrict__ stroke_count_label,
                         const float* __restrict__ sig,
                         const float* __restrict__ mlp_w1, const float* __restrict__ mlp_b1,
                         const float* __restrict__ mlp_w2, const float* __restrict__ mlp_b2,
                         const float* __restrict__ rr_w,
                         float* __restrict__ out) {
  __shared__ float row[NCLS];
  __shared__ float preds[NRAD];
  __shared__ float bv[128];
  __shared__ int   bix[128];
  __shared__ int   s_idx[5];
  __shared__ float s_val[5];
  __shared__ float s_sp[NSTRUCT];
  __shared__ float s_pn[NST];
  __shared__ int   s_stroke_pred;
  __shared__ float s_numdet;

  const int n = blockIdx.x, tid = threadIdx.x;
  const float NEG = -__builtin_inff();

#if defined(__gfx1250__) && __has_builtin(__builtin_amdgcn_global_load_async_to_lds_b32) && __has_builtin(__builtin_amdgcn_s_wait_asynccnt)
  // CDNA5 async global->LDS copy of the char-logits row.
  typedef __attribute__((address_space(1))) int* gp_i32;
  typedef __attribute__((address_space(3))) int* lp_i32;
  for (int j = tid; j < NCLS; j += 128) {
    __builtin_amdgcn_global_load_async_to_lds_b32(
        (gp_i32)(void*)(char_l + (size_t)n * NCLS + j),
        (lp_i32)(void*)(&row[j]), 0, 0);
  }
  __builtin_amdgcn_s_wait_asynccnt(0);
  __syncthreads();
  for (int j = tid; j < NCLS; j += 128)
    out[(size_t)n * NCLS + j] = row[j];            // pass-through logits
#else
  for (int j = tid; j < NCLS; j += 128) {
    float v = char_l[(size_t)n * NCLS + j];
    row[j] = v;
    out[(size_t)n * NCLS + j] = v;
  }
#endif
  for (int r = tid; r < NRAD; r += 128)            // sigmoid(x)>0.5 <=> x>0
    preds[r] = (rad_l[n * NRAD + r] > 0.f) ? 1.f : 0.f;
  __syncthreads();

  // exact top-5 (ties -> smaller index), 5 argmax passes over LDS row
  for (int k = 0; k < 5; ++k) {
    float best = NEG; int bidx = 0x7fffffff;
    for (int j = tid; j < NCLS; j += 128) {
      float v = row[j];
      if (v > best || (v == best && j < bidx)) { best = v; bidx = j; }
    }
    bv[tid] = best; bix[tid] = bidx; __syncthreads();
    for (int s = 64; s > 0; s >>= 1) {
      if (tid < s) {
        float v2 = bv[tid + s]; int i2 = bix[tid + s];
        if (v2 > bv[tid] || (v2 == bv[tid] && i2 < bix[tid])) { bv[tid] = v2; bix[tid] = i2; }
      }
      __syncthreads();
    }
    if (tid == 0) { s_idx[k] = bix[0]; s_val[k] = bv[0]; row[bix[0]] = NEG; }
    __syncthreads();
  }

  if (tid == 0) {
    // softmax over structure logits (13)
    float mx = NEG;
    for (int j = 0; j < NSTRUCT; ++j) mx = fmaxf(mx, str_l[n * NSTRUCT + j]);
    float se = 0.f, e[NSTRUCT];
    for (int j = 0; j < NSTRUCT; ++j) { e[j] = __expf(str_l[n * NSTRUCT + j] - mx); se += e[j]; }
    for (int j = 0; j < NSTRUCT; ++j) s_sp[j] = e[j] / se;
    // argmax stroke-count (first occurrence)
    int am = 0; float amv = sc_l[n * NSC + 0];
    for (int j = 1; j < NSC; ++j) { float v = sc_l[n * NSC + j]; if (v > amv) { amv = v; am = j; } }
    s_stroke_pred = am;
    // l2norm of stroke-type logits (6)
    float ss = 0.f;
    for (int j = 0; j < NST; ++j) { float v = st_l[n * NST + j]; ss += v * v; }
    float nn = fmaxf(sqrtf(ss), 1e-12f);
    for (int j = 0; j < NST; ++j) s_pn[j] = st_l[n * NST + j] / nn;
    // number of detected radicals
    float nd = 0.f;
    for (int r = 0; r < NRAD; ++r) nd += preds[r];
    s_numdet = nd;
  }
  __syncthreads();

  if (tid < 5) {
    const int k = tid, id = s_idx[k];
    const int* mrow = radical_mask + (size_t)id * NRAD;
    float det = 0.f, cnt = 0.f;
    for (int r = 0; r < NRAD; ++r) { float m = (float)mrow[r]; det += preds[r] * m; cnt += m; }
    float match_ratio = det / fmaxf(cnt, 1.f);
    float false_ratio = (s_numdet - det) / fmaxf(s_numdet, 1.f);
    float struct_match = s_sp[structure_label[id]];
    float stroke_dist = fabsf((float)(s_stroke_pred - stroke_count_label[id])) * (1.f / 29.f);
    float ss = 0.f;
    for (int j = 0; j < NST; ++j) { float v = sig[(size_t)id * NST + j]; ss += v * v; }
    float nn = fmaxf(sqrtf(ss), 1e-12f);
    float cosv = 0.f;
    for (int j = 0; j < NST; ++j) cosv += s_pn[j] * (sig[(size_t)id * NST + j] / nn);
    float f[5] = { match_ratio, false_ratio, struct_match, stroke_dist, cosv };
    float score = mlp_b2[0];
    for (int hci = 0; hci < 32; ++hci) {
      float h = mlp_b1[hci];
      for (int j2 = 0; j2 < 5; ++j2) h += mlp_w1[hci * 5 + j2] * f[j2];
      score += mlp_w2[hci] * fmaxf(h, 0.f);
    }
    out[(size_t)n * NCLS + id] = s_val[k] + rr_w[0] * score;
  }
}

// ---------------------------------------------------------------------------
// Host launcher
// ---------------------------------------------------------------------------
extern "C" void kernel_launch(void* const* d_in, const int* in_sizes, int n_in,
                              void* d_out, int out_size, void* d_ws, size_t ws_size,
                              hipStream_t stream) {
  (void)n_in; (void)out_size; (void)ws_size;

  // ---- input mapping (auto-detect flatten order via unique leaf sizes) ----
  const int x_first = (in_sizes[0] == BATCH * 64 * 64);          // x = 524288
  const int pb = x_first ? 1 : 0;
  const int psorted = (in_sizes[pb] == 64);   // sorted -> b1_beta1(64); insertion -> stem_w(576)
  static const int s_map[43] = {
    39, 38, 37,
     2,  4,  0,  3,  5,  1,
     8, 10,  6,  9, 11,  7, 14, 13, 12,
    17, 19, 15, 18, 20, 16, 23, 22, 21,
    25, 24,
    31, 30,
    34, 33,
    36, 35,
    41, 40,
    28, 26, 29, 27,
    32,
    42
  };
  enum { STEM_W = 0, STEM_G, STEM_B,
         B1_C1, B1_G1, B1_BETA1, B1_C2, B1_G2, B1_BETA2,
         B2_C1, B2_G1, B2_BETA1, B2_C2, B2_G2, B2_BETA2, B2_SKW, B2_SKG, B2_SKB,
         B3_C1, B3_G1, B3_BETA1, B3_C2, B3_G2, B3_BETA2, B3_SKW, B3_SKG, B3_SKB,
         CHAR_W, CHAR_B, RAD_W, RAD_B, SC_W, SC_B, ST_W, ST_B, STR_W, STR_B,
         MLP_W1, MLP_B1, MLP_W2, MLP_B2, RR_W, SIG };
  auto PP = [&](int slot) -> const float* {
    return (const float*)d_in[pb + (psorted ? s_map[slot] : slot)];
  };
  const float* xin; const int* rm; const int* slab; const int* sclab;
  if (x_first) {
    xin = (const float*)d_in[0];
    rm = (const int*)d_in[44]; slab = (const int*)d_in[45]; sclab = (const int*)d_in[46];
  } else {
    rm = (const int*)d_in[43]; sclab = (const int*)d_in[44]; slab = (const int*)d_in[45];
    xin = (const float*)d_in[46];
  }

  // ---- workspace bump allocator ----
  char* base = (char*)d_ws;
  size_t off = 0;
  auto alloc = [&](size_t bytes) -> void* {
    void* p = base + off;
    off += (bytes + 255) & ~(size_t)255;
    return p;
  };
  float* stem_out = (float*)alloc((size_t)33554432 * 4);   // (128,64,64,64)
  float* act1     = (float*)alloc((size_t)8388608 * 4);    // (128,64,32,32)
  float* t1a      = (float*)alloc((size_t)8388608 * 4);
  float* t2a      = (float*)alloc((size_t)8388608 * 4);
  float* act2     = (float*)alloc((size_t)2097152 * 4);    // (128,64,16,16)
  float* idn2     = (float*)alloc((size_t)4194304 * 4);    // (128,128,16,16)
  float* t1b      = (float*)alloc((size_t)4194304 * 4);
  float* t2b      = (float*)alloc((size_t)4194304 * 4);
  float* act3     = (float*)alloc((size_t)1048576 * 4);    // (128,128,8,8)
  float* idn3     = (float*)alloc((size_t)2097152 * 4);    // (128,256,8,8)
  float* t1c      = (float*)alloc((size_t)2097152 * 4);
  float* t2c      = (float*)alloc((size_t)2097152 * 4);
  float* feat     = (float*)alloc((size_t)32768 * 4);      // (128,256)
  float* char_l   = (float*)alloc((size_t)BATCH * NCLS * 4);
  float* rad_l    = (float*)alloc((size_t)BATCH * NRAD * 4);
  float* sc_l     = (float*)alloc((size_t)BATCH * NSC * 4);
  float* st_l     = (float*)alloc((size_t)BATCH * NST * 4);
  float* str_l    = (float*)alloc((size_t)BATCH * NSTRUCT * 4);
  float* stats    = (float*)alloc((size_t)512 * 4);
  __bf16* wb1c1 = (__bf16*)alloc((size_t)576 * 64 * 2);
  __bf16* wb1c2 = (__bf16*)alloc((size_t)576 * 64 * 2);
  __bf16* wb2c1 = (__bf16*)alloc((size_t)576 * 128 * 2);
  __bf16* wb2c2 = (__bf16*)alloc((size_t)1152 * 128 * 2);
  __bf16* wb2sk = (__bf16*)alloc((size_t)64 * 128 * 2);
  __bf16* wb3c1 = (__bf16*)alloc((size_t)1152 * 256 * 2);
  __bf16* wb3c2 = (__bf16*)alloc((size_t)2304 * 256 * 2);
  __bf16* wb3sk = (__bf16*)alloc((size_t)128 * 256 * 2);
  __bf16* wchar = (__bf16*)alloc((size_t)256 * NCLS * 2);
  __bf16* wrad  = (__bf16*)alloc((size_t)256 * NRAD * 2);
  __bf16* wsc   = (__bf16*)alloc((size_t)256 * NSC * 2);
  __bf16* wst   = (__bf16*)alloc((size_t)256 * NST * 2);
  __bf16* wstr  = (__bf16*)alloc((size_t)256 * NSTRUCT * 2);

  auto ilog2 = [](int v) { int s = 0; while ((1 << s) < v) ++s; return s; };
  auto cvt = [&](const float* w, __bf16* o, int total) {
    k_cvtw<<<(total + 255) / 256, 256, 0, stream>>>(w, o, total);
  };
  auto conv = [&](const float* inp, const __bf16* wb, const float* bias, float* outp,
                  int Cin, int H, int W, int Cout, int ks) {
    dim3 g((BATCH * H * W) / 128, (Cout + 63) / 64);
    int hw_s = ilog2(H * W), w_s = ilog2(W);
    if (ks == 3)
      k_gemm_wmma<3><<<g, 128, 0, stream>>>(inp, wb, bias, outp, Cin, hw_s, w_s, Cout);
    else
      k_gemm_wmma<1><<<g, 128, 0, stream>>>(inp, wb, bias, outp, Cin, hw_s, w_s, Cout);
  };
  auto bn = [&](float* t, const float* g_, const float* b_, int C, int HW,
                const float* res, int relu) {
    int lc = ilog2(C), hw_s = ilog2(HW);
    k_bn_stats<<<C, 256, 0, stream>>>(t, stats, lc, hw_s);
    int total = BATCH * C * HW;
    k_bn_apply<<<(total + 255) / 256, 256, 0, stream>>>(t, res, t, stats, g_, b_,
                                                        lc, hw_s, total, relu);
  };
  auto pool = [&](const float* inp, float* outp, int C, int H) {
    int total = BATCH * C * (H / 2) * (H / 2);
    k_maxpool<<<(total + 255) / 256, 256, 0, stream>>>(inp, outp, ilog2(H), ilog2(C), total);
  };

  // ---- weight conversions (f32 -> bf16, [Cout][K]) ----
  cvt(PP(B1_C1), wb1c1, 576 * 64);    cvt(PP(B1_C2), wb1c2, 576 * 64);
  cvt(PP(B2_C1), wb2c1, 576 * 128);   cvt(PP(B2_C2), wb2c2, 1152 * 128);
  cvt(PP(B2_SKW), wb2sk, 64 * 128);
  cvt(PP(B3_C1), wb3c1, 1152 * 256);  cvt(PP(B3_C2), wb3c2, 2304 * 256);
  cvt(PP(B3_SKW), wb3sk, 128 * 256);
  cvt(PP(CHAR_W), wchar, 256 * NCLS); cvt(PP(RAD_W), wrad, 256 * NRAD);
  cvt(PP(SC_W), wsc, 256 * NSC);      cvt(PP(ST_W), wst, 256 * NST);
  cvt(PP(STR_W), wstr, 256 * NSTRUCT);

  // ---- stem ----
  k_stem<<<(BATCH * 64 * 4096) / 256, 256, 0, stream>>>(xin, PP(STEM_W), stem_out);
  bn(stem_out, PP(STEM_G), PP(STEM_B), 64, 4096, nullptr, 1);
  pool(stem_out, act1, 64, 64);

  // ---- block 1 (64->64 @32x32) ----
  conv(act1, wb1c1, nullptr, t1a, 64, 32, 32, 64, 3);
  bn(t1a, PP(B1_G1), PP(B1_BETA1), 64, 1024, nullptr, 1);
  conv(t1a, wb1c2, nullptr, t2a, 64, 32, 32, 64, 3);
  bn(t2a, PP(B1_G2), PP(B1_BETA2), 64, 1024, act1, 1);
  pool(t2a, act2, 64, 32);

  // ---- block 2 (64->128 @16x16, skip 1x1) ----
  conv(act2, wb2sk, nullptr, idn2, 64, 16, 16, 128, 1);
  bn(idn2, PP(B2_SKG), PP(B2_SKB), 128, 256, nullptr, 0);
  conv(act2, wb2c1, nullptr, t1b, 64, 16, 16, 128, 3);
  bn(t1b, PP(B2_G1), PP(B2_BETA1), 128, 256, nullptr, 1);
  conv(t1b, wb2c2, nullptr, t2b, 128, 16, 16, 128, 3);
  bn(t2b, PP(B2_G2), PP(B2_BETA2), 128, 256, idn2, 1);
  pool(t2b, act3, 128, 16);

  // ---- block 3 (128->256 @8x8, skip 1x1) ----
  conv(act3, wb3sk, nullptr, idn3, 128, 8, 8, 256, 1);
  bn(idn3, PP(B3_SKG), PP(B3_SKB), 256, 64, nullptr, 0);
  conv(act3, wb3c1, nullptr, t1c, 128, 8, 8, 256, 3);
  bn(t1c, PP(B3_G1), PP(B3_BETA1), 256, 64, nullptr, 1);
  conv(t1c, wb3c2, nullptr, t2c, 256, 8, 8, 256, 3);
  bn(t2c, PP(B3_G2), PP(B3_BETA2), 256, 64, idn3, 1);

  // ---- GAP + heads (1x1 "convs" on WMMA) ----
  k_gap<<<(32768 + 255) / 256, 256, 0, stream>>>(t2c, feat);
  conv(feat, wchar, PP(CHAR_B), char_l, 256, 1, 1, NCLS, 1);
  conv(feat, wrad,  PP(RAD_B),  rad_l,  256, 1, 1, NRAD, 1);
  conv(feat, wsc,   PP(SC_B),   sc_l,   256, 1, 1, NSC, 1);
  conv(feat, wst,   PP(ST_B),   st_l,   256, 1, 1, NST, 1);
  conv(feat, wstr,  PP(STR_B),  str_l,  256, 1, 1, NSTRUCT, 1);

  // ---- symbolic reranker + scatter into d_out ----
  k_rerank<<<BATCH, 128, 0, stream>>>(char_l, rad_l, sc_l, st_l, str_l,
      rm, slab, sclab, PP(SIG), PP(MLP_W1), PP(MLP_B1), PP(MLP_W2), PP(MLP_B2),
      PP(RR_W), (float*)d_out);
}
